// MoEMLP_14577119003273
// MI455X (gfx1250) — matrix operand order repaired and checked by reading, not compiled, not run on hardware
//
#include <hip/hip_runtime.h>
#include <stdint.h>

typedef __attribute__((ext_vector_type(16))) __bf16 v16bf;
typedef __attribute__((ext_vector_type(8)))  float  v8f;

#define T_TOK   4096
#define D_DIM   1024
#define E_NUM   8
#define HID_DIM 4096
#define TPAD    (T_TOK + 32*E_NUM)   /* 4352 sorted rows, buckets 32-aligned */
#define MTILES  (TPAD/32)            /* 136 32-token tiles                   */
#define CHUNK   512                  /* K elements staged in LDS per pass    */

// ---------------------------------------------------------------------------
// init: order = -1, zero counters (must run every launch: graph replay)
// ---------------------------------------------------------------------------
__global__ void init_kernel(int* __restrict__ order, int* __restrict__ counts,
                            int* __restrict__ cursors, float* __restrict__ psum) {
    int i = blockIdx.x * 256 + threadIdx.x;
    if (i < TPAD) order[i] = -1;
    if (i < E_NUM) { counts[i] = 0; cursors[i] = 0; psum[i] = 0.0f; }
}

// ---------------------------------------------------------------------------
// router: one wave per token; logits -> softmax probs, argmax, counts
// ---------------------------------------------------------------------------
__global__ __launch_bounds__(256)
void router_kernel(const float* __restrict__ x, const float* __restrict__ wr,
                   int* __restrict__ expert_idx, int* __restrict__ counts,
                   float* __restrict__ probs) {
    const int token = blockIdx.x * 8 + (threadIdx.x >> 5);
    const int lane  = threadIdx.x & 31;
    if (token >= T_TOK) return;
    float acc[E_NUM];
#pragma unroll
    for (int e = 0; e < E_NUM; ++e) acc[e] = 0.0f;
    const float* xp = x + (size_t)token * D_DIM;
    for (int k = lane; k < D_DIM; k += 32) {
        float xv = xp[k];
#pragma unroll
        for (int e = 0; e < E_NUM; ++e) acc[e] = fmaf(xv, wr[e * D_DIM + k], acc[e]);
    }
#pragma unroll
    for (int off = 16; off > 0; off >>= 1)
#pragma unroll
        for (int e = 0; e < E_NUM; ++e) acc[e] += __shfl_xor(acc[e], off, 32);
    if (lane == 0) {
        float mx = acc[0]; int am = 0;
#pragma unroll
        for (int e = 1; e < E_NUM; ++e) if (acc[e] > mx) { mx = acc[e]; am = e; }
        float ex[E_NUM]; float den = 0.0f;
#pragma unroll
        for (int e = 0; e < E_NUM; ++e) { ex[e] = expf(acc[e] - mx); den += ex[e]; }
        float inv = 1.0f / den;
#pragma unroll
        for (int e = 0; e < E_NUM; ++e) probs[(size_t)token * E_NUM + e] = ex[e] * inv;
        expert_idx[token] = am;
        atomicAdd(&counts[am], 1);
    }
}

// ---------------------------------------------------------------------------
// offsets: 32-aligned per-expert bucket starts
// ---------------------------------------------------------------------------
__global__ void offsets_kernel(const int* __restrict__ counts, int* __restrict__ poff) {
    if (threadIdx.x == 0 && blockIdx.x == 0) {
        int o = 0; poff[0] = 0;
        for (int e = 0; e < E_NUM; ++e) { o += (counts[e] + 31) & ~31; poff[e + 1] = o; }
    }
}

__global__ void scatter_kernel(const int* __restrict__ expert_idx,
                               const int* __restrict__ poff, int* __restrict__ cursors,
                               int* __restrict__ order) {
    int t = blockIdx.x * 256 + threadIdx.x;
    if (t < T_TOK) {
        int e = expert_idx[t];
        int p = atomicAdd(&cursors[e], 1);
        order[poff[e] + p] = t;
    }
}

// ---------------------------------------------------------------------------
// build dense bf16 weights, pre-swizzled into the WMMA B-matrix lane layout:
// Wb[((e*NOUT/16 + ntile)*NK/32 + ktile)*512 + L*16 + j]
//   = W[ntile*16 + L%16][ktile*32 + 16*(L/16) + j]
//   = sum_i A[e,i,j,k] * S[e,i,o,l]     (O=j*s_out+o, I=k*s_in+l)
// ---------------------------------------------------------------------------
__global__ __launch_bounds__(256)
void build_weights(const float* __restrict__ A, const float* __restrict__ S,
                   __bf16* __restrict__ Wb, int NOUT, int NK,
                   int so_sh, int si_sh) {
    long long tid = (long long)blockIdx.x * 256 + threadIdx.x;
    int j  = (int)(tid & 15);
    int L  = (int)((tid >> 4) & 31);
    long long t2 = tid >> 9;
    int nk32 = NK >> 5;
    int kt = (int)(t2 % nk32);
    long long t3 = t2 / nk32;
    int nt16 = NOUT >> 4;
    int ntl = (int)(t3 % nt16);
    int e   = (int)(t3 / nt16);
    int O = ntl * 16 + (L & 15);
    int I = kt * 32 + ((L >> 4) << 4) + j;
    int jj = O >> so_sh, o = O & ((1 << so_sh) - 1);
    int kk = I >> si_sh, l = I & ((1 << si_sh) - 1);
    int s_out = 1 << so_sh, s_in = 1 << si_sh;
    float w = 0.0f;
#pragma unroll
    for (int i = 0; i < 4; ++i)
        w = fmaf(A[((e * 4 + i) * 4 + jj) * 4 + kk],
                 S[((size_t)(e * 4 + i) * s_out + o) * s_in + l], w);
    Wb[tid] = (__bf16)w;
}

// ---------------------------------------------------------------------------
// WMMA GEMM over expert-sorted 32-token tiles.
//   block = 256 thr = 8 waves; tile = 32 rows x 512 cols.
//   wave  = 2 M-subtiles x 4 N-tiles = 8 accumulators; each B fragment is
//   reused by both M-subtiles -> ~32 FLOP per L2 byte of weight traffic.
//   FC:  A = gathered x (fp32->bf16), epilogue leaky(0.5)^2 -> hsq (bf16)
//   PROJ:A = hsq (bf16, already sorted), epilogue scatter fp32 to out[token]
// ---------------------------------------------------------------------------
template <int K_TOTAL, bool IS_FC>
__global__ __launch_bounds__(256)
void phm_gemm(const float* __restrict__ x, const __bf16* __restrict__ actb,
              const __bf16* __restrict__ Wb, const int* __restrict__ order,
              const int* __restrict__ poff, __bf16* __restrict__ hsq_out,
              float* __restrict__ out, int NOUT) {
    const int base = blockIdx.x * 32;
    if (base >= poff[E_NUM]) return;
    const int tid  = threadIdx.x;
    const int lane = tid & 31;
    const int wave = tid >> 5;
    const int m    = lane & 15;
    const int half = lane >> 4;

    // expert owning this tile (buckets are 32-aligned)
    int e = 0;
#pragma unroll
    for (int i = 0; i < E_NUM; ++i) if (base >= poff[i + 1]) e = i + 1;

    __shared__ __bf16 lA[32][CHUNK + 8];   // +8 pad: 4-bank stagger per row
    __shared__ int tok[32];
    if (tid < 32) tok[tid] = order[base + tid];
    __syncthreads();

    v8f acc[2][4] = {};

    for (int kc = 0; kc < K_TOTAL; kc += CHUNK) {
        // ---- stage 32 x CHUNK A-panel into LDS (bf16) ----
        if constexpr (IS_FC) {
#pragma unroll 4
            for (int it = 0; it < 32 * CHUNK / (256 * 4); ++it) {
                int f = (tid + 256 * it) << 2;
                int r = f >> 9, c = f & (CHUNK - 1);
                int t = tok[r];
                float4 v = make_float4(0.f, 0.f, 0.f, 0.f);
                if (t >= 0) v = *(const float4*)(x + (size_t)t * K_TOTAL + kc + c);
                lA[r][c + 0] = (__bf16)v.x; lA[r][c + 1] = (__bf16)v.y;
                lA[r][c + 2] = (__bf16)v.z; lA[r][c + 3] = (__bf16)v.w;
            }
        } else {
#pragma unroll 4
            for (int it = 0; it < 32 * CHUNK / (256 * 8); ++it) {
                int f = (tid + 256 * it) << 3;
                int r = f >> 9, c = f & (CHUNK - 1);
                *(uint4*)&lA[r][c] =
                    *(const uint4*)(actb + (size_t)(base + r) * K_TOTAL + kc + c);
            }
        }
        __syncthreads();

        // ---- CHUNK/32 k-tiles of v_wmma_f32_16x16x32_bf16 ----
        for (int ktl = 0; ktl < CHUNK / 32; ++ktl) {
            union AU { v16bf v; uint4 q[2]; } a[2];
            const int kb = ktl * 32;
#pragma unroll
            for (int mt = 0; mt < 2; ++mt) {
                a[mt].q[0] = *(const uint4*)&lA[mt * 16 + m][kb + 16 * half];
                a[mt].q[1] = *(const uint4*)&lA[mt * 16 + m][kb + 16 * half + 8];
            }
            const int ktg = (kc >> 5) + ktl;
#pragma unroll
            for (int an = 0; an < 4; ++an) {
                int ntile = blockIdx.y * 32 + wave * 4 + an;
                const __bf16* bp = Wb +
                    ((((size_t)e * (NOUT >> 4) + ntile) * (K_TOTAL >> 5) + ktg) << 9) +
                    (size_t)lane * 16;
                union { v16bf v; uint4 q[2]; } b;
                b.q[0] = *(const uint4*)(bp);
                b.q[1] = *(const uint4*)(bp + 8);
#pragma unroll
                for (int mt = 0; mt < 2; ++mt)   // B reused across both M-subtiles
                    acc[mt][an] = __builtin_amdgcn_wmma_f32_16x16x32_bf16(
                        false, a[mt].v, false, b.v, (short)0, acc[mt][an], false, false);
            }
        }
        __syncthreads();
    }

    // ---- epilogue: D frag element (r,lane) is row r+8*half, col lane%16 ----
#pragma unroll
    for (int mt = 0; mt < 2; ++mt) {
#pragma unroll
        for (int an = 0; an < 4; ++an) {
            int ncol = (blockIdx.y * 32 + wave * 4 + an) * 16 + m;
#pragma unroll
            for (int r = 0; r < 8; ++r) {
                int mrow = mt * 16 + r + 8 * half;
                float v = acc[mt][an][r];
                if constexpr (IS_FC) {
                    float h = (v >= 0.0f) ? v : 0.5f * v;   // leaky relu slope 0.5
                    hsq_out[(size_t)(base + mrow) * NOUT + ncol] = (__bf16)(h * h);
                } else {
                    int t = tok[mrow];
                    if (t >= 0) out[(size_t)t * NOUT + ncol] = v;
                }
            }
        }
    }
}

// ---------------------------------------------------------------------------
// aux loss: deterministic fixed-order reductions
// ---------------------------------------------------------------------------
__global__ void reduce_probs(const float* __restrict__ probs, float* __restrict__ psum) {
    const int e = blockIdx.x;
    __shared__ float s[256];
    float a = 0.0f;
    for (int t = threadIdx.x; t < T_TOK; t += 256) a += probs[(size_t)t * E_NUM + e];
    s[threadIdx.x] = a; __syncthreads();
    for (int o = 128; o > 0; o >>= 1) {
        if (threadIdx.x < o) s[threadIdx.x] += s[threadIdx.x + o];
        __syncthreads();
    }
    if (threadIdx.x == 0) psum[e] = s[0];
}

__global__ void finalize_kernel(const int* __restrict__ counts,
                                const float* __restrict__ psum,
                                float* __restrict__ aux_out) {
    if (threadIdx.x == 0 && blockIdx.x == 0) {
        float a = 0.0f;
        for (int e = 0; e < E_NUM; ++e)
            a += ((float)counts[e] / (float)T_TOK) * (psum[e] / (float)T_TOK);
        aux_out[0] = a * (float)E_NUM;
    }
}

// ---------------------------------------------------------------------------
extern "C" void kernel_launch(void* const* d_in, const int* in_sizes, int n_in,
                              void* d_out, int out_size, void* d_ws, size_t ws_size,
                              hipStream_t stream) {
    const float* x      = (const float*)d_in[0];
    const float* wr     = (const float*)d_in[1];
    const float* A_fc   = (const float*)d_in[2];
    const float* S_fc   = (const float*)d_in[3];
    const float* A_proj = (const float*)d_in[4];
    const float* S_proj = (const float*)d_in[5];
    float* out = (float*)d_out;
    float* aux = out + (size_t)T_TOK * D_DIM;

    char* w = (char*)d_ws;
    auto take = [&](size_t bytes, size_t align) -> char* {
        uintptr_t q = ((uintptr_t)w + align - 1) & ~(uintptr_t)(align - 1);
        w = (char*)(q + bytes);
        return (char*)q;
    };
    int*    expert_idx = (int*)   take((size_t)T_TOK * 4, 256);
    int*    counts     = (int*)   take(E_NUM * 4, 64);
    int*    cursors    = (int*)   take(E_NUM * 4, 64);
    int*    poff       = (int*)   take((E_NUM + 1) * 4, 64);
    float*  psum       = (float*) take(E_NUM * 4, 64);
    int*    order      = (int*)   take((size_t)TPAD * 4, 256);
    float*  probs      = (float*) take((size_t)T_TOK * E_NUM * 4, 256);
    __bf16* hsq        = (__bf16*)take((size_t)TPAD * HID_DIM * 2, 256);
    __bf16* wfc        = (__bf16*)take((size_t)E_NUM * HID_DIM * D_DIM * 2, 256);
    __bf16* wproj      = (__bf16*)take((size_t)E_NUM * D_DIM * HID_DIM * 2, 256);

    init_kernel<<<(TPAD + 255) / 256, 256, 0, stream>>>(order, counts, cursors, psum);
    router_kernel<<<T_TOK / 8, 256, 0, stream>>>(x, wr, expert_idx, counts, probs);
    offsets_kernel<<<1, 64, 0, stream>>>(counts, poff);
    scatter_kernel<<<T_TOK / 256, 256, 0, stream>>>(expert_idx, poff, cursors, order);

    const long long wfc_elems = (long long)E_NUM * HID_DIM * D_DIM;
    build_weights<<<(unsigned)(wfc_elems / 256), 256, 0, stream>>>(
        A_fc, S_fc, wfc, HID_DIM, D_DIM, /*so=log2(1024)*/10, /*si=log2(256)*/8);
    build_weights<<<(unsigned)(wfc_elems / 256), 256, 0, stream>>>(
        A_proj, S_proj, wproj, D_DIM, HID_DIM, /*so=log2(256)*/8, /*si=log2(1024)*/10);

    phm_gemm<D_DIM, true><<<dim3(MTILES, HID_DIM / 512), 256, 0, stream>>>(
        x, nullptr, wfc, order, poff, hsq, nullptr, HID_DIM);
    phm_gemm<HID_DIM, false><<<dim3(MTILES, D_DIM / 512), 256, 0, stream>>>(
        nullptr, hsq, wproj, order, poff, nullptr, out, D_DIM);

    reduce_probs<<<E_NUM, 256, 0, stream>>>(probs, psum);
    finalize_kernel<<<1, 64, 0, stream>>>(counts, psum, aux);
}
